// Mamba_1812476199078
// MI455X (gfx1250) — compile-verified, matrix operand order checked by reading, not run on hardware
//
#include <hip/hip_runtime.h>
#include <hip/hip_bf16.h>

// ---------------------------------------------------------------------------
// Mamba-2D visual SSM pipeline for MI455X (gfx1250, wave32, WMMA)
// B=2, DM=128, DI=256, L=4096, DS=16, DC=4, DTR=8
// All f16 GEMM operands stored K-contiguous (weights M x K, activations N x K)
// so every WMMA fragment loads as two 16-byte v8h chunks.
// ---------------------------------------------------------------------------

typedef __attribute__((ext_vector_type(16))) _Float16 v16h;
typedef __attribute__((ext_vector_type(8)))  _Float16 v8h;
typedef __attribute__((ext_vector_type(8)))  float    v8f;

#define Bb   2
#define DMc  128
#define DIc  256
#define Lc   4096
#define DSc  16

__device__ __forceinline__ float sigf(float x) { return 1.0f / (1.0f + __expf(-x)); }

// -------------------------------------------------------------- conversions
__global__ void cvt_f2h(const float* __restrict__ s, _Float16* __restrict__ d, int n) {
    int i = blockIdx.x * blockDim.x + threadIdx.x;
    if (i < n) d[i] = (_Float16)s[i];
}

__global__ void cvt_f2h_pad_rows(const float* __restrict__ s, _Float16* __restrict__ d,
                                 int rows, int padRows, int cols) {
    int i = blockIdx.x * blockDim.x + threadIdx.x;
    if (i >= padRows * cols) return;
    int r = i / cols;
    d[i] = (r < rows) ? (_Float16)s[i] : (_Float16)0.0f;
}

// src: (Bc, Kdim, Ndim) f32  ->  dst: (Bc, Ndim, Kdim) f16   (Kdim = pow2)
__global__ void cvt_f2h_T(const float* __restrict__ s, _Float16* __restrict__ d,
                          int Kdim, int Ndim) {
    int idx = blockIdx.x * blockDim.x + threadIdx.x;  // over Bc*Ndim*Kdim (dst order)
    int k = idx & (Kdim - 1);
    int n = (idx / Kdim) % Ndim;
    int b = idx / (Kdim * Ndim);
    d[idx] = (_Float16)s[((long long)b * Kdim + k) * Ndim + n];
}

// ---------------------------------------------------- generic f16 WMMA GEMM
// C[b] (MxN, f32) = A (MxK f16, row-major, shared) * Bt[b] (NxK f16, row-major)
// One wave computes MT vertically-stacked 16x16 tiles, reusing one B fragment
// across MT WMMAs. K compile-time -> fully unrolled back-to-back v_wmma chain.
// 8 waves (256 thr) per block, each wave its own n-tile.
template <int K, int MT>
__global__ void gemm_f16_wmma(const _Float16* __restrict__ A,
                              const _Float16* __restrict__ Bt,
                              float* __restrict__ C,
                              int N, long long bStride, long long cStride) {
    const int bIdx = blockIdx.z;
    const _Float16* Bp = Bt + (long long)bIdx * bStride;
    float* Cp = C + (long long)bIdx * cStride;

    const int lane = threadIdx.x & 31;
    const int wave = threadIdx.x >> 5;
    const int mBase0 = blockIdx.x * (16 * MT);
    const int nBase = (blockIdx.y * 8 + wave) * 16;

    const int mn = lane & 15;   // row (A) / col (B) index within tile
    const int h  = lane >> 4;   // half-wave select

    v8f acc[MT];
    #pragma unroll
    for (int t = 0; t < MT; ++t) acc[t] = (v8f){};

    const _Float16* bRow = Bp + (long long)(nBase + mn) * K;

    #pragma unroll
    for (int k0 = 0; k0 < K; k0 += 32) {
        __builtin_prefetch(bRow + k0 + 32, 0, 1);
        // 16-bit fragment layout: elems 0..7 -> K=8h+j ; elems 8..15 -> K=16+8h+j
        v8h blo = *(const v8h*)(bRow + k0 + 8 * h);
        v8h bhi = *(const v8h*)(bRow + k0 + 16 + 8 * h);
        v16h bv;
        #pragma unroll
        for (int j = 0; j < 8; ++j) { bv[j] = blo[j]; bv[8 + j] = bhi[j]; }
        #pragma unroll
        for (int t = 0; t < MT; ++t) {
            const _Float16* aRow = A + (long long)(mBase0 + t * 16 + mn) * K;
            v8h alo = *(const v8h*)(aRow + k0 + 8 * h);
            v8h ahi = *(const v8h*)(aRow + k0 + 16 + 8 * h);
            v16h a;
            #pragma unroll
            for (int j = 0; j < 8; ++j) { a[j] = alo[j]; a[8 + j] = ahi[j]; }
            acc[t] = __builtin_amdgcn_wmma_f32_16x16x32_f16(
                false, a, false, bv, (short)0, acc[t], false, false);
        }
    }
    // C/D layout: lane n = lane&15 ; VGPR v holds row v + 8h
    #pragma unroll
    for (int t = 0; t < MT; ++t) {
        #pragma unroll
        for (int v = 0; v < 8; ++v) {
            int row = mBase0 + t * 16 + v + 8 * h;
            Cp[(long long)row * N + nBase + mn] = acc[t][v];
        }
    }
}

// ------------------------------------------------------- depthwise 5x5 conv
__global__ void dwconv5x5(const float* __restrict__ in, const float* __restrict__ w,
                          float* __restrict__ out) {
    int idx = blockIdx.x * blockDim.x + threadIdx.x;  // b*512*4096 total
    int x = idx & 63, y = (idx >> 6) & 63, c = (idx >> 12) & 511, b = idx >> 21;
    const float* wp = w + c * 25;
    const float* ip = in + (((long long)(b * 512 + c)) << 12);
    float acc = 0.0f;
    #pragma unroll
    for (int ky = 0; ky < 5; ++ky) {
        int yy = y + ky - 2;
        if (yy < 0 || yy > 63) continue;
        #pragma unroll
        for (int kx = 0; kx < 5; ++kx) {
            int xx = x + kx - 2;
            if (xx < 0 || xx > 63) continue;
            acc += wp[ky * 5 + kx] * ip[(yy << 6) + xx];
        }
    }
    out[idx] = acc;
}

// ---------------------------------------------------- channel attention path
__global__ void xstats(const float* __restrict__ xz1, float* __restrict__ avg,
                       float* __restrict__ mx) {
    int b = blockIdx.x >> 8, d = blockIdx.x & 255;
    const float* p = xz1 + ((long long)(b * 512 + d)) * Lc;
    float s = 0.0f, m = -1e30f;
    for (int l = threadIdx.x; l < Lc; l += 256) { float v = p[l]; s += v; m = fmaxf(m, v); }
    __shared__ float ss[256], sm[256];
    ss[threadIdx.x] = s; sm[threadIdx.x] = m;
    __syncthreads();
    for (int o = 128; o > 0; o >>= 1) {
        if (threadIdx.x < o) {
            ss[threadIdx.x] += ss[threadIdx.x + o];
            sm[threadIdx.x] = fmaxf(sm[threadIdx.x], sm[threadIdx.x + o]);
        }
        __syncthreads();
    }
    if (threadIdx.x == 0) { avg[blockIdx.x] = ss[0] * (1.0f / Lc); mx[blockIdx.x] = sm[0]; }
}

__global__ void ca_mlp(const float* __restrict__ avg, const float* __restrict__ mx,
                       const float* __restrict__ fc1, const float* __restrict__ fc2,
                       float* __restrict__ scale) {
    int b = blockIdx.x, t = threadIdx.x;
    __shared__ float hA[16], hM[16];
    if (t < 32) {
        int r = t & 15;
        const float* src = (t < 16) ? (avg + b * 256) : (mx + b * 256);
        float s = 0.0f;
        for (int d = 0; d < 256; ++d) s += fc1[r * 256 + d] * src[d];
        float sv = s * sigf(s);                 // silu
        if (t < 16) hA[r] = sv; else hM[r] = sv;
    }
    __syncthreads();
    float s = 0.0f;
    #pragma unroll
    for (int r = 0; r < 16; ++r) s += fc2[t * 16 + r] * (hA[r] + hM[r]);
    scale[b * 256 + t] = sigf(s);
}

__global__ void apply_ca(const float* __restrict__ xz1, const float* __restrict__ scale,
                         float* __restrict__ xca) {
    int idx = blockIdx.x * blockDim.x + threadIdx.x;  // b*256*4096
    int l = idx & 4095, d = (idx >> 12) & 255, b = idx >> 20;
    xca[idx] = xz1[(((long long)(b * 512 + d)) << 12) + l] * scale[b * 256 + d];
}

// ---------------------------------------------------- spatial attention path
__global__ void zstats(const float* __restrict__ xz1, float* __restrict__ stats) {
    int idx = blockIdx.x * blockDim.x + threadIdx.x;  // b*4096
    int l = idx & 4095, b = idx >> 12;
    const float* p = xz1 + ((long long)(b * 512 + 256)) * Lc + l;
    float s = 0.0f, m = -1e30f;
    for (int d = 0; d < 256; ++d) { float v = p[(long long)d * Lc]; s += v; m = fmaxf(m, v); }
    stats[(b * 2 + 0) * Lc + l] = s * (1.0f / 256.0f);
    stats[(b * 2 + 1) * Lc + l] = m;
}

__global__ void saconv(const float* __restrict__ stats, const float* __restrict__ w,
                       float* __restrict__ att) {
    int idx = blockIdx.x * blockDim.x + threadIdx.x;  // b*4096
    int l = idx & 4095, b = idx >> 12;
    float a = 0.0f;
    #pragma unroll
    for (int i = 0; i < 2; ++i)
        #pragma unroll
        for (int j = 0; j < 7; ++j) {
            int ll = l + j - 3;
            if (ll < 0 || ll > 4095) continue;
            a += w[i * 7 + j] * stats[(b * 2 + i) * Lc + ll];
        }
    att[idx] = a;
}

// ------------------------------------------- causal depthwise conv1d + SiLU
// f32 result in channel-major (b,d,l); f16 copy transposed to (b,l,d)
__global__ void conv1d_silu(const float* __restrict__ xca, const float* __restrict__ w,
                            const float* __restrict__ bias, float* __restrict__ xo,
                            _Float16* __restrict__ xhT) {
    int idx = blockIdx.x * blockDim.x + threadIdx.x;  // b*256*4096
    int l = idx & 4095, d = (idx >> 12) & 255, b = idx >> 20;
    const float* p = xca + (idx - l);
    float acc = bias[d];
    #pragma unroll
    for (int j = 0; j < 4; ++j) {
        int ll = l - 3 + j;
        if (ll >= 0) acc += w[d * 4 + j] * p[ll];
    }
    float v = acc * sigf(acc);
    xo[idx] = v;
    xhT[((long long)b * Lc + l) * 256 + d] = (_Float16)v;
}

// ------------------------------------------------------- delta = softplus()
__global__ void dt_softplus(const float* __restrict__ xdbl, const float* __restrict__ wdt,
                            const float* __restrict__ bdt, float* __restrict__ delta) {
    int idx = blockIdx.x * blockDim.x + threadIdx.x;  // b*256*4096
    int l = idx & 4095, d = (idx >> 12) & 255, b = idx >> 20;
    const float* p = xdbl + (long long)b * 48 * Lc + l;
    float s = bdt[d];
    #pragma unroll
    for (int r = 0; r < 8; ++r) s += wdt[d * 8 + r] * p[(long long)r * Lc];
    delta[idx] = (s > 20.0f) ? s : log1pf(__expf(s));
}

// ---------------------------------------------------- selective scan + gate
// one block per batch, one lane per channel d; B/C broadcast via LDS per step.
// y written (b,l,d) f16 -> coalesced across lanes AND K-contiguous for out_proj.
__global__ void scan_kernel(const float* __restrict__ xdbl, const float* __restrict__ delta,
                            const float* __restrict__ xconv, const float* __restrict__ A_log,
                            const float* __restrict__ Dp, const float* __restrict__ xz1,
                            const float* __restrict__ att, _Float16* __restrict__ yhT) {
    int b = blockIdx.x, d = threadIdx.x;
    float negA[DSc], s[DSc];
    #pragma unroll
    for (int n = 0; n < DSc; ++n) { negA[n] = -__expf(A_log[d * DSc + n]); s[n] = 0.0f; }
    float Dv = Dp[d];
    const float* xb = xdbl + (long long)b * 48 * Lc;
    const float* dl = delta + ((long long)(b * 256 + d)) * Lc;
    const float* xv = xconv + ((long long)(b * 256 + d)) * Lc;
    const float* zr = xz1 + ((long long)(b * 512 + 256 + d)) * Lc;
    const float* ar = att + (long long)b * Lc;
    _Float16* yo = yhT + (long long)b * Lc * 256 + d;
    __shared__ float Bs[DSc], Cs[DSc];

    for (int l = 0; l < Lc; ++l) {
        if (d < 16)      Bs[d]      = xb[(8 + d) * Lc + l];
        else if (d < 32) Cs[d - 16] = xb[(24 + (d - 16)) * Lc + l];
        __syncthreads();
        float dd = dl[l], xx = xv[l];
        float y = 0.0f;
        #pragma unroll
        for (int n = 0; n < DSc; ++n) {
            s[n] = __expf(dd * negA[n]) * s[n] + dd * Bs[n] * xx;
            y += s[n] * Cs[n];
        }
        y += xx * Dv;
        float zf = zr[l] * sigf(ar[l]);     // spatial-attention gate
        float gate = zf * sigf(zf);         // silu(z)
        yo[(long long)l * 256] = (_Float16)(y * gate);
        __syncthreads();
    }
}

// ---------------------------------------------------------------------------
extern "C" void kernel_launch(void* const* d_in, const int* in_sizes, int n_in,
                              void* d_out, int out_size, void* d_ws, size_t ws_size,
                              hipStream_t stream) {
    const float* hs      = (const float*)d_in[0];   // (2,128,64,64)
    const float* w_in    = (const float*)d_in[1];   // (512,128)
    const float* w_dw    = (const float*)d_in[2];   // (512,1,5,5)
    const float* w_c1d   = (const float*)d_in[3];   // (256,1,4)
    const float* b_c1d   = (const float*)d_in[4];   // (256)
    const float* w_xp    = (const float*)d_in[5];   // (40,256)
    const float* w_dt    = (const float*)d_in[6];   // (256,8)
    const float* b_dt    = (const float*)d_in[7];   // (256)
    const float* A_log   = (const float*)d_in[8];   // (256,16)
    const float* Dvec    = (const float*)d_in[9];   // (256)
    const float* w_out   = (const float*)d_in[10];  // (128,256)
    const float* w_fc1   = (const float*)d_in[11];  // (16,256)
    const float* w_fc2   = (const float*)d_in[12];  // (256,16)
    const float* w_sa    = (const float*)d_in[13];  // (1,2,7)
    float* out = (float*)d_out;                     // (2,128,64,64)

    char* ws = (char*)d_ws;
    size_t off = 0;
    auto alloc = [&](size_t bytes) { size_t r = off; off += (bytes + 255) & ~(size_t)255; return r; };
    float*    xz0   = (float*)(ws + alloc((size_t)Bb * 512 * Lc * 4));
    float*    xz1   = (float*)(ws + alloc((size_t)Bb * 512 * Lc * 4));
    float*    xca   = (float*)(ws + alloc((size_t)Bb * 256 * Lc * 4));
    float*    xconv = (float*)(ws + alloc((size_t)Bb * 256 * Lc * 4));
    float*    delta = (float*)(ws + alloc((size_t)Bb * 256 * Lc * 4));
    float*    xdbl  = (float*)(ws + alloc((size_t)Bb * 48 * Lc * 4));
    float*    stats = (float*)(ws + alloc((size_t)Bb * 2 * Lc * 4));
    float*    att   = (float*)(ws + alloc((size_t)Bb * Lc * 4));
    float*    avg   = (float*)(ws + alloc((size_t)Bb * 256 * 4));
    float*    mx    = (float*)(ws + alloc((size_t)Bb * 256 * 4));
    float*    scale = (float*)(ws + alloc((size_t)Bb * 256 * 4));
    _Float16* hWin  = (_Float16*)(ws + alloc((size_t)512 * 128 * 2));
    _Float16* hHSt  = (_Float16*)(ws + alloc((size_t)Bb * Lc * 128 * 2));  // (b,l,k)
    _Float16* hWx   = (_Float16*)(ws + alloc((size_t)48 * 256 * 2));
    _Float16* hWo   = (_Float16*)(ws + alloc((size_t)128 * 256 * 2));
    _Float16* hXcT  = (_Float16*)(ws + alloc((size_t)Bb * Lc * 256 * 2));  // (b,l,d)
    _Float16* hYT   = (_Float16*)(ws + alloc((size_t)Bb * Lc * 256 * 2));  // (b,l,d)

    const int T = 256;
    auto grid1 = [&](int n) { return dim3((unsigned)((n + T - 1) / T)); };

    // 0) f32 -> f16 prep (weights row-major MxK; activations transposed NxK)
    cvt_f2h<<<grid1(512 * 128), T, 0, stream>>>(w_in, hWin, 512 * 128);
    cvt_f2h_T<<<grid1(Bb * Lc * 128), T, 0, stream>>>(hs, hHSt, 128, Lc);
    cvt_f2h<<<grid1(128 * 256), T, 0, stream>>>(w_out, hWo, 128 * 256);
    cvt_f2h_pad_rows<<<grid1(48 * 256), T, 0, stream>>>(w_xp, hWx, 40, 48, 256);

    // 1) in_proj: (512x128) @ (128x4096) per batch -> xz0   [MT=4, K=128]
    gemm_f16_wmma<128, 4><<<dim3(512 / 64, Lc / 128, Bb), T, 0, stream>>>(
        hWin, hHSt, xz0, Lc, (long long)Lc * 128, (long long)512 * Lc);

    // 2) depthwise 5x5 conv -> xz1
    dwconv5x5<<<grid1(Bb * 512 * Lc), T, 0, stream>>>(xz0, w_dw, xz1);

    // 3) channel attention on x
    xstats<<<dim3(Bb * 256), T, 0, stream>>>(xz1, avg, mx);
    ca_mlp<<<dim3(Bb), T, 0, stream>>>(avg, mx, w_fc1, w_fc2, scale);
    apply_ca<<<grid1(Bb * 256 * Lc), T, 0, stream>>>(xz1, scale, xca);

    // 4) spatial attention stats on z
    zstats<<<grid1(Bb * Lc), T, 0, stream>>>(xz1, stats);
    saconv<<<grid1(Bb * Lc), T, 0, stream>>>(stats, w_sa, att);

    // 5) causal depthwise conv1d + bias + silu -> xconv (f32) + hXcT (f16, (b,l,d))
    conv1d_silu<<<grid1(Bb * 256 * Lc), T, 0, stream>>>(xca, w_c1d, b_c1d, xconv, hXcT);

    // 6) x_proj: (48x256 padded) @ (256x4096) per batch -> xdbl   [MT=3, K=256]
    gemm_f16_wmma<256, 3><<<dim3(1, Lc / 128, Bb), T, 0, stream>>>(
        hWx, hXcT, xdbl, Lc, (long long)Lc * 256, (long long)48 * Lc);

    // 7) delta = softplus(dt_proj @ dt_lo + b)
    dt_softplus<<<grid1(Bb * 256 * Lc), T, 0, stream>>>(xdbl, w_dt, b_dt, delta);

    // 8) selective scan + D skip + gating -> hYT (f16, (b,l,d))
    scan_kernel<<<dim3(Bb), T, 0, stream>>>(xdbl, delta, xconv, A_log, Dvec, xz1, att, hYT);

    // 9) out_proj: (128x256) @ (256x4096) per batch -> out (b,m,l)   [MT=4, K=256]
    gemm_f16_wmma<256, 4><<<dim3(128 / 64, Lc / 128, Bb), T, 0, stream>>>(
        hWo, hYT, out, Lc, (long long)Lc * 256, (long long)128 * Lc);
}
